// MeshConv_8323646619907
// MI455X (gfx1250) — matrix operand order repaired and checked by reading, not compiled, not run on hardware
//
#include <hip/hip_runtime.h>
#include <hip/hip_bf16.h>

typedef __bf16 bf16;
typedef __attribute__((ext_vector_type(16))) __bf16        v16bf;
typedef __attribute__((ext_vector_type(8)))  float         v8f;
typedef __attribute__((ext_vector_type(4)))  unsigned int  v4u;

#define KDIM     320      // 5*C feature width
#define OUTCH    128
#define ROWS     64       // edges per block
#define FSTRIDE  328      // bf16 elems per feat row (320 + 8 pad) -> bank-conflict-free b128

__device__ __forceinline__ float4 f4min(float4 a, float4 b) {
    return make_float4(fminf(a.x,b.x), fminf(a.y,b.y), fminf(a.z,b.z), fminf(a.w,b.w));
}
__device__ __forceinline__ float4 f4max(float4 a, float4 b) {
    return make_float4(fmaxf(a.x,b.x), fmaxf(a.y,b.y), fmaxf(a.z,b.z), fmaxf(a.w,b.w));
}
__device__ __forceinline__ void st4bf(bf16* p, float4 v) {
    p[0] = (bf16)v.x; p[1] = (bf16)v.y; p[2] = (bf16)v.z; p[3] = (bf16)v.w;
}

// ---------------- kernel 0: W f32 -> bf16 ----------------
__global__ __launch_bounds__(256)
void prep_kernel(const float* __restrict__ W, bf16* __restrict__ Wbf, int n) {
    int i = blockIdx.x * 256 + threadIdx.x;
    if (i < n) Wbf[i] = (bf16)W[i];
}

// ---------------- kernel 1: gather + minmax + GEMM (bf16 WMMA, f32 acc) ----------------
__global__ __launch_bounds__(256)
void gemm_kernel(const float* __restrict__ x, const long long* __restrict__ nb,
                 const bf16* __restrict__ Wbf, float* __restrict__ out, int E)
{
    __shared__ bf16 feat[ROWS * FSTRIDE];   // 64 * 328 * 2 = 41984 B

    const int tid = threadIdx.x;
    const long long base = (long long)blockIdx.x * ROWS;

    // ---- stage feature tile: 4 threads per row, 16 channels each ----
    {
        const int row = tid >> 2;        // 0..63
        const int seg = tid & 3;         // 16-channel chunk
        const long long e = base + row;
        bf16* frow = &feat[row * FSTRIDE + seg * 16];
        if (e < (long long)E) {
            long long i0 = nb[e*4+0], i1 = nb[e*4+1], i2 = nb[e*4+2], i3 = nb[e*4+3];
            const long long Em1 = (long long)E - 1;
            i0 = i0 < 0 ? 0 : (i0 > Em1 ? Em1 : i0);
            i1 = i1 < 0 ? 0 : (i1 > Em1 ? Em1 : i1);
            i2 = i2 < 0 ? 0 : (i2 > Em1 ? Em1 : i2);
            i3 = i3 < 0 ? 0 : (i3 > Em1 ? Em1 : i3);
            const float4* xs = (const float4*)(x + e *64 + seg*16);
            const float4* n0 = (const float4*)(x + i0*64 + seg*16);
            const float4* n1 = (const float4*)(x + i1*64 + seg*16);
            const float4* n2 = (const float4*)(x + i2*64 + seg*16);
            const float4* n3 = (const float4*)(x + i3*64 + seg*16);
            #pragma unroll
            for (int j = 0; j < 4; ++j) {
                float4 a  = xs[j];
                float4 b0 = n0[j], b1 = n1[j], b2 = n2[j], b3 = n3[j];
                st4bf(frow + 0*64 + j*4, a);
                st4bf(frow + 1*64 + j*4, f4min(b0, b1));
                st4bf(frow + 2*64 + j*4, f4max(b0, b1));
                st4bf(frow + 3*64 + j*4, f4min(b2, b3));
                st4bf(frow + 4*64 + j*4, f4max(b2, b3));
            }
        } else {
            #pragma unroll
            for (int s = 0; s < 5; ++s)
                #pragma unroll
                for (int j = 0; j < 16; ++j) frow[s*64 + j] = (bf16)0.0f;
        }
    }
    __syncthreads();

    // ---- compute: wave w owns output channels [w*16, w*16+16) ----
    const int wave = tid >> 5;
    const int lane = tid & 31;
    const int half = lane >> 4;     // K-half selector (ISA 7.12.2 layouts)
    const int nl   = lane & 15;     // row (A) / column (B,D) within tile
    const int nch  = wave * 16 + nl;

    v8f acc[4] = {v8f{}, v8f{}, v8f{}, v8f{}};
    const bf16* wrow = Wbf + nch * KDIM;

    #pragma unroll
    for (int kt = 0; kt < 10; ++kt) {
        // B fragment: 16 contiguous bf16 of W row nch, K = kt*32 + half*16 ..+15
        union { v4u u[2]; v16bf v; } B;
        const v4u* wp = (const v4u*)(wrow + kt*32 + half*16);
        B.u[0] = wp[0];
        B.u[1] = wp[1];
        #pragma unroll
        for (int mt = 0; mt < 4; ++mt) {
            // A fragment: row nl of this M-tile; K = kt*32 + half*16 (+0..15)
            union { v4u u[2]; v16bf v; } A;
            const bf16* arow = &feat[(mt*16 + nl) * FSTRIDE + kt*32];
            A.u[0] = *(const v4u*)(arow + half*16);
            A.u[1] = *(const v4u*)(arow + half*16 + 8);
            acc[mt] = __builtin_amdgcn_wmma_f32_16x16x32_bf16(
                false, A.v, false, B.v, (short)0, acc[mt], false, false);
        }
    }

    // ---- store D: VGPR r -> row (r + half*8), col nl ----
    #pragma unroll
    for (int mt = 0; mt < 4; ++mt) {
        const long long row0 = base + mt*16 + half*8;
        #pragma unroll
        for (int r = 0; r < 8; ++r) {
            const long long e = row0 + r;
            if (e < (long long)E) out[e * OUTCH + nch] = acc[mt][r];
        }
    }
}

// ---------------- kernel 2: per-channel partial sums (coalesced, deterministic) ----------------
__global__ __launch_bounds__(256)
void stats_partial(const float* __restrict__ h, float* __restrict__ psum,
                   float* __restrict__ psq, int E)
{
    const int tid  = threadIdx.x;
    const int c    = tid & 127;
    const int slot = tid >> 7;                 // 0 or 1
    float s = 0.0f, s2 = 0.0f;
    for (long long r = blockIdx.x * 2 + slot; r < (long long)E; r += 512) {
        float v = h[r * OUTCH + c];
        s += v; s2 += v * v;
    }
    __shared__ float ls[256], ls2[256];
    ls[tid] = s; ls2[tid] = s2;
    __syncthreads();
    if (tid < 128) {
        psum[blockIdx.x * 128 + tid] = ls[tid]  + ls[tid + 128];
        psq [blockIdx.x * 128 + tid] = ls2[tid] + ls2[tid + 128];
    }
}

// ---------------- kernel 3: final BN stats -> scale/shift ----------------
__global__ __launch_bounds__(128)
void stats_final(const float* __restrict__ psum, const float* __restrict__ psq,
                 const float* __restrict__ gamma, const float* __restrict__ beta,
                 float* __restrict__ scale, float* __restrict__ shift, int E)
{
    const int c = threadIdx.x;
    float s = 0.0f, s2 = 0.0f;
    for (int b = 0; b < 256; ++b) { s += psum[b*128 + c]; s2 += psq[b*128 + c]; }
    const float inv  = 1.0f / (float)E;
    const float mean = s * inv;
    const float var  = s2 * inv - mean * mean;
    const float sc   = gamma[c] * rsqrtf(var + 1e-5f);
    scale[c] = sc;
    shift[c] = beta[c] - mean * sc;
}

// ---------------- kernel 4: normalize + ReLU in place ----------------
__global__ __launch_bounds__(256)
void finalize_kernel(float* __restrict__ h, const float* __restrict__ scale,
                     const float* __restrict__ shift, long long n4)
{
    const float4* sc4 = (const float4*)scale;
    const float4* sh4 = (const float4*)shift;
    float4* h4 = (float4*)h;
    long long i = (long long)blockIdx.x * 256 + threadIdx.x;
    const long long stride = (long long)gridDim.x * 256;
    for (; i < n4; i += stride) {
        const int c4 = (int)(i & 31);           // 32 float4 per 128-ch row
        float4 v = h4[i];
        const float4 s = sc4[c4];
        const float4 b = sh4[c4];
        v.x = fmaxf(0.0f, v.x * s.x + b.x);
        v.y = fmaxf(0.0f, v.y * s.y + b.y);
        v.z = fmaxf(0.0f, v.z * s.z + b.z);
        v.w = fmaxf(0.0f, v.w * s.w + b.w);
        h4[i] = v;
    }
}

extern "C" void kernel_launch(void* const* d_in, const int* in_sizes, int n_in,
                              void* d_out, int out_size, void* d_ws, size_t ws_size,
                              hipStream_t stream)
{
    const float*     x     = (const float*)d_in[0];
    const long long* nb    = (const long long*)d_in[1];   // int64 per reference
    const float*     W     = (const float*)d_in[2];
    const float*     gamma = (const float*)d_in[3];
    const float*     beta  = (const float*)d_in[4];
    float*           out   = (float*)d_out;

    const int E = in_sizes[0] / 64;           // x is (E, 64)

    // workspace layout
    char*  ws    = (char*)d_ws;
    bf16*  Wbf   = (bf16*)ws;                               //  81920 B
    float* psum  = (float*)(ws + 81920);                    // 131072 B (256 blk * 128 ch)
    float* psq   = (float*)(ws + 81920 + 131072);           // 131072 B
    float* scale = (float*)(ws + 81920 + 262144);           //    512 B
    float* shift = scale + 128;                             //    512 B

    prep_kernel <<<(OUTCH*KDIM + 255)/256, 256, 0, stream>>>(W, Wbf, OUTCH*KDIM);
    gemm_kernel <<<(E + ROWS - 1)/ROWS,   256, 0, stream>>>(x, nb, Wbf, out, E);
    stats_partial<<<256, 256, 0, stream>>>(out, psum, psq, E);
    stats_final <<<1, 128, 0, stream>>>(psum, psq, gamma, beta, scale, shift, E);
    finalize_kernel<<<4096, 256, 0, stream>>>(out, scale, shift, (long long)E * (OUTCH/4));
}
